// Seq2SeqClassifier_66718021976404
// MI455X (gfx1250) — matrix operand order repaired and checked by reading, not compiled, not run on hardware
//
#include <hip/hip_runtime.h>
#include <hip/hip_bf16.h>

// ---------------------------------------------------------------------------
// Seq2Seq LSTM encoder/decoder for MI455X (gfx1250).
//
// Batch-1 matvecs are weight-bandwidth bound (all weights fit the 192MB L2),
// so: bf16-pack every weight matrix into the per-lane B-fragment layout of
// v_wmma_f32_16x16x32_bf16, run a persistent 1024-thread workgroup with
// LDS-resident state, and stream weight fragments (2x global_load_b128 +
// global_prefetch + 1 wmma per 16x32 tile). A-fragments hold the activation
// vector in matrix row 0; rows 1..15 are DON'T-CARE (their D rows are never
// read), so every lane loads the row-0 fragment unconditionally (LDS
// broadcast) -- no EXEC masking in the inner loop. Two K-chunks are processed
// per iteration in separate register buffers so B loads overlap the WMMAs;
// the loop is kept at its manual 2x pipelining (#pragma unroll 1) so the
// compiler cannot over-unroll, hoist A loads, and spill them to scratch.
// ---------------------------------------------------------------------------

#define TLEN   128
#define MAXL   4
#define ENC_H  512
#define DEC_H  1024
#define NLAB   500
#define EMB_E  300
#define EMB_LE 100

typedef __attribute__((ext_vector_type(16))) __bf16 v16bf;
typedef __attribute__((ext_vector_type(8)))  float  v8f;

union Frag {
    unsigned short s[16];
    uint4          q[2];
    v16bf          v;
};

__device__ __host__ inline unsigned short f2bf_bits(float f) {
    union { float f; unsigned u; } cv; cv.f = f;
    unsigned u = cv.u;
    unsigned r = u + 0x7FFFu + ((u >> 16) & 1u);   // round-to-nearest-even
    return (unsigned short)(r >> 16);
}

__device__ inline float sigf(float x) { return 1.f / (1.f + expf(-x)); }

// A-pack slot mapping for one 32-value K-chunk (contiguous, matches B pack):
//   lane0  frag elems {0..15} <- k {0..15}
//   lane16 frag elems {0..15} <- k {16..31}
// For k-offset r in [0,32): half = r>>4 (0=lane0 frag, 1=lane16 frag),
//                           e    = r&15
__device__ inline void apack_slot(int r, int& half, int& e) {
    half = (r >> 4) & 1;
    e    = r & 15;
}

// Accumulate one packed weight segment:
// acc[0,n] += sum_k x[k] * W[nt*16+n, k]  over Ktiles chunks of 32.
// ap points at this segment's first A-fragment for this lane half
// (32 ushorts per chunk: [0..15]=lane0 frag, [16..31]=lane16 frag).
// All lanes load the row-0 fragment; A rows 1..15 are don't-care.
__device__ inline void mv_seg(v8f& acc, const unsigned short* __restrict__ pw,
                              int Ktiles, int nt,
                              const unsigned short* __restrict__ ap,
                              int lane) {
    const uint4* bp  = (const uint4*)(pw + (long)nt * Ktiles * 512 + lane * 16);
    const uint4* ap4 = (const uint4*)ap;
    int kt = 0;
#pragma unroll 1
    for (; kt + 2 <= Ktiles; kt += 2) {
        Frag a0, b0, a1, b1;
        a0.q[0] = ap4[0];  a0.q[1] = ap4[1];
        b0.q[0] = bp[0];   b0.q[1] = bp[1];
        a1.q[0] = ap4[4];  a1.q[1] = ap4[5];
        b1.q[0] = bp[64];  b1.q[1] = bp[65];
        __builtin_prefetch(bp + 128, 0, 1);
        acc = __builtin_amdgcn_wmma_f32_16x16x32_bf16(
            false, a0.v, false, b0.v, (short)0, acc, false, false);
        acc = __builtin_amdgcn_wmma_f32_16x16x32_bf16(
            false, a1.v, false, b1.v, (short)0, acc, false, false);
        bp  += 128;   // 2 x 512 ushorts (two 32x16 B fragments)
        ap4 += 8;     // 2 x 32 ushorts
    }
    if (kt < Ktiles) {
        Frag a, b;
        a.q[0] = ap4[0]; a.q[1] = ap4[1];
        b.q[0] = bp[0];  b.q[1] = bp[1];
        acc = __builtin_amdgcn_wmma_f32_16x16x32_bf16(
            false, a.v, false, b.v, (short)0, acc, false, false);
    }
}

// ---------------------------------------------------------------------------
// Prep kernels (wide, parallel)
// ---------------------------------------------------------------------------

// Pack fp32 W[N,K] row-major into bf16 WMMA-B fragments:
// out[((nt*Ktiles+kt)*32 + lane)*16 + e] = W[nt*16 + (lane&15)][kt*32 + (lane>>4)*16 + e]
__global__ void s2s_pack_w_bf16(const float* __restrict__ W, int N, int K,
                                int Ktiles, int Ntiles,
                                unsigned short* __restrict__ out) {
    long total = (long)Ntiles * Ktiles * 512;
    for (long o = (long)blockIdx.x * blockDim.x + threadIdx.x; o < total;
         o += (long)gridDim.x * blockDim.x) {
        int e    = (int)(o & 15);
        int lane = (int)((o >> 4) & 31);
        long ft  = o >> 9;                    // nt*Ktiles + kt
        int kt   = (int)(ft % Ktiles);
        int nt   = (int)(ft / Ktiles);
        int n = nt * 16 + (lane & 15);
        int k = kt * 32 + ((lane >> 4) << 4) + e;
        float v = (n < N && k < K) ? W[(long)n * K + k] : 0.f;
        out[o] = f2bf_bits(v);
    }
}

// X0[t][k] = enc_emb_w[inputs[t]][k] for k<300 else 0 (fp32, row stride 1024)
__global__ void s2s_embed_enc(const int* __restrict__ inputs,
                              const float* __restrict__ emb,
                              float* __restrict__ X0) {
    int idx = blockIdx.x * blockDim.x + threadIdx.x;
    if (idx >= TLEN * 1024) return;
    int t = idx >> 10, k = idx & 1023;
    X0[idx] = (k < EMB_E) ? emb[(long)inputs[t] * EMB_E + k] : 0.f;
}

// scores init: NEG everywhere, 0 at label index 0
__global__ void s2s_init_scores(float* __restrict__ out) {
    int idx = blockIdx.x * blockDim.x + threadIdx.x;
    if (idx >= TLEN * MAXL * NLAB) return;
    out[idx] = (idx % NLAB == 0) ? 0.f : -1e10f;
}

// ---------------------------------------------------------------------------
// Main persistent kernel: 1 block x 1024 threads (32 waves on one WGP)
// ---------------------------------------------------------------------------

struct S2SParams {
    const int*   inputs;
    const int*   input_lengths;
    const int*   gold;
    const float* dec_emb;
    const float* enc_bih[2][2];   // [layer][dir] dir0=fwd
    const float* enc_bhh[2][2];
    const float* dec_bih[2];
    const float* dec_bhh[2];
    const float* out_b;
    const unsigned short* enc_wih[2][2];
    const unsigned short* enc_whh[2][2];
    const unsigned short* dec_wih[2];
    const unsigned short* dec_whh[2];
    const unsigned short* out_w;
    const float* X0;              // [T,1024] embeddings (padded)
    float*       X1;              // [T,1024] encoder layer-0 output
    float*       hcsave;          // [4096] encoder final h (2x1024) + c (2x1024)
    float*       out;             // [T,MAXL,NLAB]
};

__global__ void __launch_bounds__(1024, 1) s2s_main(S2SParams P) {
    __shared__ float hbuf[2][1024];
    __shared__ float cbuf[2][1024];
    __shared__ float gbuf[4096];
    __shared__ float dsbuf[512];
    __shared__ float elab[128];
    __shared__ __align__(64) unsigned short apack[2][64][32];
    __shared__ int   s_ti, s_li, s_pred;

    const int tid   = threadIdx.x;
    const int lane  = tid & 31;
    const int wave  = tid >> 5;
    const int dir   = wave >> 4;            // 0 = fwd waves, 1 = bwd waves
    const int dwave = wave & 15;
    const int dtid  = tid - dir * 512;      // 0..511 within direction
    const int asel  = ((lane >> 4) & 1) * 16;   // lane-half offset into A-pack
    const int tok_len = P.input_lengths[0];

    // ---------------- Encoder: 2 layers, fwd/bwd in lockstep --------------
    for (int layer = 0; layer < 2; ++layer) {
        const float* Xin = (layer == 0) ? P.X0 : P.X1;
        const int KtX = (layer == 0) ? 10 : 32;    // ceil(din/32): 300->10, 1024->32
        const int nch = KtX + 16;                  // + Whh chunks (512/32)
        const unsigned short* Wih = P.enc_wih[layer][dir];
        const unsigned short* Whh = P.enc_whh[layer][dir];
        const float* bih = P.enc_bih[layer][dir];
        const float* bhh = P.enc_bhh[layer][dir];

        hbuf[dir][dtid] = 0.f;
        cbuf[dir][dtid] = 0.f;
        __syncthreads();

        for (int s = 0; s < TLEN; ++s) {
            const int t = (dir == 0) ? s : (TLEN - 1 - s);

            // A-pack: chunks [0,KtX) from x_t (global, zero-padded rows),
            //         chunks [KtX,nch) from h (LDS)
            for (int v = dtid; v < nch * 32; v += 512) {
                int c = v >> 5, r = v & 31, half, e;
                apack_slot(r, half, e);
                float val = (c < KtX) ? Xin[(long)t * 1024 + c * 32 + r]
                                      : hbuf[dir][(c - KtX) * 32 + r];
                apack[dir][c][half * 16 + e] = f2bf_bits(val);
            }
            __syncthreads();

            // gates: 128 tiles of 16 rows per direction, 8 tiles per wave
#pragma unroll 1
            for (int tt = 0; tt < 8; ++tt) {
                const int nt = dwave * 8 + tt;
                v8f acc = {0.f, 0.f, 0.f, 0.f, 0.f, 0.f, 0.f, 0.f};
                mv_seg(acc, Wih, KtX, nt, &apack[dir][0][asel], lane);
                mv_seg(acc, Whh, 16,  nt, &apack[dir][KtX][asel], lane);
                if (lane < 16) gbuf[dir * 2048 + nt * 16 + lane] = acc[0];
            }
            __syncthreads();

            // pointwise LSTM cell (512 threads per direction)
            {
                const float* g = gbuf + dir * 2048;
                const int j = dtid;
                float gi = g[j]        + bih[j]        + bhh[j];
                float gf = g[512 + j]  + bih[512 + j]  + bhh[512 + j];
                float gg = g[1024 + j] + bih[1024 + j] + bhh[1024 + j];
                float go = g[1536 + j] + bih[1536 + j] + bhh[1536 + j];
                float c = sigf(gf) * cbuf[dir][j] + sigf(gi) * tanhf(gg);
                float h = sigf(go) * tanhf(c);
                cbuf[dir][j] = c;
                hbuf[dir][j] = h;
                if (layer == 0)   // layer-1 output is unused downstream
                    P.X1[(long)t * 1024 + dir * 512 + j] = h;
            }
            __syncthreads();
        }
        // save final states to workspace: concat(fwd, bwd) per layer
        P.hcsave[layer * 1024 + dir * 512 + dtid]        = hbuf[dir][dtid];
        P.hcsave[2048 + layer * 1024 + dir * 512 + dtid] = cbuf[dir][dtid];
        __threadfence();
        __syncthreads();
    }

    // ---------------- Decoder init ----------------------------------------
    hbuf[0][tid] = P.hcsave[tid];         cbuf[0][tid] = P.hcsave[2048 + tid];
    hbuf[1][tid] = P.hcsave[1024 + tid];  cbuf[1][tid] = P.hcsave[3072 + tid];
    if (tid < EMB_LE) elab[tid] = P.dec_emb[1 * EMB_LE + tid];   // SOS = 1
    if (tid == 0) { s_ti = 0; s_li = 0; }
    __syncthreads();

    // ---------------- Decoder loop (T*MAXL steps, teacher forced) ---------
    for (int it = 0; it < TLEN * MAXL; ++it) {
        const int ti = s_ti, li = s_li;
        if (ti >= tok_len) break;            // state frozen afterwards
        const int tclip = (ti < TLEN - 1) ? ti : (TLEN - 1);

        // ---- layer 0: chunks [0,13) = xv = concat(elab, embed, pad),
        //               chunks [13,45) = h0
        for (int v = tid; v < 45 * 32; v += 1024) {
            int c = v >> 5, r = v & 31, half, e;
            apack_slot(r, half, e);
            float val;
            if (c < 13) {
                int k = c * 32 + r;
                val = (k < EMB_LE) ? elab[k]
                    : (k < 400)    ? P.X0[(long)tclip * 1024 + (k - EMB_LE)]
                                   : 0.f;
            } else {
                val = hbuf[0][(c - 13) * 32 + r];
            }
            apack[0][c][half * 16 + e] = f2bf_bits(val);
        }
        __syncthreads();
#pragma unroll 1
        for (int tt = 0; tt < 8; ++tt) {
            const int nt = wave * 8 + tt;
            v8f acc = {0.f, 0.f, 0.f, 0.f, 0.f, 0.f, 0.f, 0.f};
            mv_seg(acc, P.dec_wih[0], 13, nt, &apack[0][0][asel],  lane);
            mv_seg(acc, P.dec_whh[0], 32, nt, &apack[0][13][asel], lane);
            if (lane < 16) gbuf[nt * 16 + lane] = acc[0];
        }
        __syncthreads();
        {
            const int j = tid;
            const float* bih = P.dec_bih[0];
            const float* bhh = P.dec_bhh[0];
            float gi = gbuf[j]        + bih[j]        + bhh[j];
            float gf = gbuf[1024 + j] + bih[1024 + j] + bhh[1024 + j];
            float gg = gbuf[2048 + j] + bih[2048 + j] + bhh[2048 + j];
            float go = gbuf[3072 + j] + bih[3072 + j] + bhh[3072 + j];
            float c = sigf(gf) * cbuf[0][j] + sigf(gi) * tanhf(gg);
            float h = sigf(go) * tanhf(c);
            cbuf[0][j] = c;
            hbuf[0][j] = h;
        }
        __syncthreads();

        // ---- layer 1: chunks [0,32) = h0 (new), chunks [32,64) = h1 (old)
        for (int v = tid; v < 64 * 32; v += 1024) {
            int c = v >> 5, r = v & 31, half, e;
            apack_slot(r, half, e);
            float val = (c < 32) ? hbuf[0][c * 32 + r] : hbuf[1][(c - 32) * 32 + r];
            apack[0][c][half * 16 + e] = f2bf_bits(val);
        }
        __syncthreads();
#pragma unroll 1
        for (int tt = 0; tt < 8; ++tt) {
            const int nt = wave * 8 + tt;
            v8f acc = {0.f, 0.f, 0.f, 0.f, 0.f, 0.f, 0.f, 0.f};
            mv_seg(acc, P.dec_wih[1], 32, nt, &apack[0][0][asel],  lane);
            mv_seg(acc, P.dec_whh[1], 32, nt, &apack[0][32][asel], lane);
            if (lane < 16) gbuf[nt * 16 + lane] = acc[0];
        }
        __syncthreads();
        {
            const int j = tid;
            const float* bih = P.dec_bih[1];
            const float* bhh = P.dec_bhh[1];
            float gi = gbuf[j]        + bih[j]        + bhh[j];
            float gf = gbuf[1024 + j] + bih[1024 + j] + bhh[1024 + j];
            float gg = gbuf[2048 + j] + bih[2048 + j] + bhh[2048 + j];
            float go = gbuf[3072 + j] + bih[3072 + j] + bhh[3072 + j];
            float c = sigf(gf) * cbuf[1][j] + sigf(gi) * tanhf(gg);
            float h = sigf(go) * tanhf(c);
            cbuf[1][j] = c;
            hbuf[1][j] = h;
        }
        __syncthreads();

        // ---- output projection: pack new h1 (32 chunks), 1 tile per wave
        {
            int c = tid >> 5, r = tid & 31, half, e;
            apack_slot(r, half, e);
            apack[1][c][half * 16 + e] = f2bf_bits(hbuf[1][c * 32 + r]);
        }
        __syncthreads();
        {
            const int nt = wave;
            v8f acc = {0.f, 0.f, 0.f, 0.f, 0.f, 0.f, 0.f, 0.f};
            mv_seg(acc, P.out_w, 32, nt, &apack[1][0][asel], lane);
            if (lane < 16) {
                const int n = nt * 16 + lane;
                if (n < NLAB) dsbuf[n] = acc[0] + P.out_b[n];
            }
        }
        __syncthreads();

        // commit scores for (tclip, li)
        if (tid < NLAB)
            P.out[((long)tclip * MAXL + li) * NLAB + tid] = dsbuf[tid];

        // teacher-forcing control
        if (tid == 0) {
            int pred = P.gold[tclip * MAXL + li];
            int adv  = (pred == 2) || (li == MAXL - 1);   // EOT or last slot
            s_pred = pred;
            s_ti   = ti + adv;
            s_li   = adv ? 0 : (li + 1);
        }
        __syncthreads();
        if (tid < EMB_LE) elab[tid] = P.dec_emb[(long)s_pred * EMB_LE + tid];
        __syncthreads();
    }
}

// ---------------------------------------------------------------------------
// Host launch
// ---------------------------------------------------------------------------

extern "C" void kernel_launch(void* const* d_in, const int* in_sizes, int n_in,
                              void* d_out, int out_size, void* d_ws, size_t ws_size,
                              hipStream_t stream) {
    (void)in_sizes; (void)out_size; (void)ws_size;
    if (n_in < 31) return;

    // Input order: top-level setup_inputs() dict order; nested pytrees in
    // jax sorted-key order ('bwd'<'fwd'; 'Whh'<'Wih'<'bhh'<'bih').
    const int*   inputs  = (const int*)d_in[0];
    const int*   lens    = (const int*)d_in[1];
    const int*   gold    = (const int*)d_in[2];
    const float* enc_emb = (const float*)d_in[3];
    const float* dec_emb = (const float*)d_in[4];
    const float* eWhh[2][2], *eWih[2][2], *ebhh[2][2], *ebih[2][2];
    for (int l = 0; l < 2; ++l) {
        int b = 5 + l * 8;
        eWhh[l][1] = (const float*)d_in[b + 0];  // bwd
        eWih[l][1] = (const float*)d_in[b + 1];
        ebhh[l][1] = (const float*)d_in[b + 2];
        ebih[l][1] = (const float*)d_in[b + 3];
        eWhh[l][0] = (const float*)d_in[b + 4];  // fwd
        eWih[l][0] = (const float*)d_in[b + 5];
        ebhh[l][0] = (const float*)d_in[b + 6];
        ebih[l][0] = (const float*)d_in[b + 7];
    }
    const float* dWhh[2], *dWih[2], *dbhh[2], *dbih[2];
    for (int l = 0; l < 2; ++l) {
        int b = 21 + l * 4;
        dWhh[l] = (const float*)d_in[b + 0];
        dWih[l] = (const float*)d_in[b + 1];
        dbhh[l] = (const float*)d_in[b + 2];
        dbih[l] = (const float*)d_in[b + 3];
    }
    const float* out_W = (const float*)d_in[29];
    const float* out_b = (const float*)d_in[30];

    // workspace layout (256B aligned chunks)
    char* ws = (char*)d_ws;
    size_t cur = 0;
    auto alloc = [&](size_t bytes) -> void* {
        void* p = ws + cur;
        cur = (cur + bytes + 255) & ~(size_t)255;
        return p;
    };
    float* X0     = (float*)alloc((size_t)TLEN * 1024 * 4);
    float* X1     = (float*)alloc((size_t)TLEN * 1024 * 4);
    float* hcsave = (float*)alloc((size_t)4096 * 4);

    auto pack = [&](const float* W, int N, int K, int Kt, int Nt) -> unsigned short* {
        unsigned short* p = (unsigned short*)alloc((size_t)Nt * Kt * 512 * 2);
        long total = (long)Nt * Kt * 512;
        int grid = (int)((total + 255) / 256);
        if (grid > 4096) grid = 4096;
        s2s_pack_w_bf16<<<grid, 256, 0, stream>>>(W, N, K, Kt, Nt, p);
        return p;
    };

    S2SParams P;
    P.inputs = inputs; P.input_lengths = lens; P.gold = gold;
    P.dec_emb = dec_emb; P.out_b = out_b;
    for (int l = 0; l < 2; ++l)
        for (int d = 0; d < 2; ++d) {
            int dinK = (l == 0) ? EMB_E : 2 * ENC_H;
            int KtX  = (l == 0) ? 10 : 32;
            P.enc_wih[l][d] = pack(eWih[l][d], 4 * ENC_H, dinK, KtX, 128);
            P.enc_whh[l][d] = pack(eWhh[l][d], 4 * ENC_H, ENC_H, 16, 128);
            P.enc_bih[l][d] = ebih[l][d];
            P.enc_bhh[l][d] = ebhh[l][d];
        }
    P.dec_wih[0] = pack(dWih[0], 4 * DEC_H, EMB_LE + EMB_E, 13, 256);
    P.dec_whh[0] = pack(dWhh[0], 4 * DEC_H, DEC_H, 32, 256);
    P.dec_wih[1] = pack(dWih[1], 4 * DEC_H, DEC_H, 32, 256);
    P.dec_whh[1] = pack(dWhh[1], 4 * DEC_H, DEC_H, 32, 256);
    for (int l = 0; l < 2; ++l) { P.dec_bih[l] = dbih[l]; P.dec_bhh[l] = dbhh[l]; }
    P.out_w = pack(out_W, NLAB, DEC_H, 32, 32);
    P.X0 = X0; P.X1 = X1; P.hcsave = hcsave; P.out = (float*)d_out;

    s2s_embed_enc<<<(TLEN * 1024 + 255) / 256, 256, 0, stream>>>(inputs, enc_emb, X0);
    s2s_init_scores<<<(TLEN * MAXL * NLAB + 255) / 256, 256, 0, stream>>>((float*)d_out);
    s2s_main<<<1, 1024, 0, stream>>>(P);
}
